// TemporalEncoder_22368189677717
// MI455X (gfx1250) — matrix operand order repaired and checked by reading, not compiled, not run on hardware
//
#include <hip/hip_runtime.h>
#include <math.h>

// ---------------------------------------------------------------------------
// Problem constants (match reference)
// ---------------------------------------------------------------------------
constexpr int Bc  = 8;
constexpr int Tc  = 1536;
constexpr int Dc  = 512;
constexpr int Hc  = 8;
constexpr int HDc = 64;
constexpr int Lc  = 4;
constexpr int Mrows = Bc * Tc;          // 12288 token rows

typedef __bf16 v16bf __attribute__((ext_vector_type(16)));
typedef __bf16 v8bf  __attribute__((ext_vector_type(8)));
typedef float  v8f   __attribute__((ext_vector_type(8)));

union v16bf_u { v16bf v; v8bf h[2]; };

// ---------------------------------------------------------------------------
// fp32 -> bf16 conversion (weights)
// ---------------------------------------------------------------------------
__global__ void cvt_bf16_kernel(const float* __restrict__ in,
                                __bf16* __restrict__ out, long n) {
    long i = (long)blockIdx.x * 256 + threadIdx.x;
    if (i < n) out[i] = (__bf16)in[i];
}

// ---------------------------------------------------------------------------
// Depthwise conv k=5 pad=2 (fp32 math), emits bf16 for the pointwise GEMM
// ---------------------------------------------------------------------------
__global__ void dwconv_kernel(const float* __restrict__ x,
                              const float* __restrict__ w,   // (5,1,D) flat
                              const float* __restrict__ bias,
                              __bf16* __restrict__ out) {
    long i = (long)blockIdx.x * 256 + threadIdx.x;   // < B*T*D
    int d = (int)(i & (Dc - 1));
    int t = (int)((i / Dc) % Tc);
    float acc = bias[d];
#pragma unroll
    for (int k = 0; k < 5; ++k) {
        int tt = t + k - 2;
        if (tt >= 0 && tt < Tc)
            acc += x[i + (long)(k - 2) * Dc] * w[k * Dc + d];
    }
    out[i] = (__bf16)acc;
}

// ---------------------------------------------------------------------------
// Sinusoidal positional encoding, added in-place to fp32 residual stream
// ---------------------------------------------------------------------------
__global__ void pe_add_kernel(float* __restrict__ xw) {
    long i = (long)blockIdx.x * 256 + threadIdx.x;   // < B*T*D
    int d = (int)(i & (Dc - 1));
    int t = (int)((i / Dc) % Tc);
    float ang = (float)t * __expf(-(float)(d & ~1) * (9.210340371976184f / (float)Dc));
    xw[i] += (d & 1) ? cosf(ang) : sinf(ang);
}

// ---------------------------------------------------------------------------
// LayerNorm: one wave32 per row, fp32 in -> bf16 out
// ---------------------------------------------------------------------------
__global__ __launch_bounds__(256) void ln_kernel(const float* __restrict__ in,
                                                 const float* __restrict__ g,
                                                 const float* __restrict__ b,
                                                 __bf16* __restrict__ out) {
    int row  = blockIdx.x * 8 + (threadIdx.x >> 5);
    int lane = threadIdx.x & 31;
    const float* p = in + (long)row * Dc;
    float s = 0.f, ss = 0.f;
    for (int i = lane; i < Dc; i += 32) { float v = p[i]; s += v; ss += v * v; }
#pragma unroll
    for (int off = 16; off; off >>= 1) {
        s  += __shfl_xor(s,  off, 32);
        ss += __shfl_xor(ss, off, 32);
    }
    float mean = s / (float)Dc;
    float var  = ss / (float)Dc - mean * mean;
    float r    = rsqrtf(var + 1e-5f);
    __bf16* o = out + (long)row * Dc;
    for (int i = lane; i < Dc; i += 32)
        o[i] = (__bf16)((p[i] - mean) * r * g[i] + b[i]);
}

// ---------------------------------------------------------------------------
// WMMA bf16 GEMM:  C[M,N] = A[M,K] * W[N,K]^T  (+bias, gelu, residual)
// Block = 256 thr = 8 waves -> 64(M) x 64(N) tile.
// Wave = 16(M) x 32(N): one A fragment, two B fragments, two WMMAs per k-step.
// A tile staged with async global->LDS (ASYNCcnt), TRIPLE-buffered LDS,
// one barrier per k-step, branch-free steady-state loop (tail peeled).
// ---------------------------------------------------------------------------
__global__ __launch_bounds__(256) void gemm_bf16_kernel(
    const __bf16* __restrict__ A,      // [M,K] bf16
    const __bf16* __restrict__ W,      // [N,K] bf16 (row-major weight)
    const float*  __restrict__ bias,   // [N] or null
    const float*  __restrict__ resIn,  // [M,N] fp32 or null
    float*        __restrict__ outF,   // [M,N] fp32 or null
    __bf16*       __restrict__ outB,   // [M,N] bf16 or null
    int M, int N, int K, int doGelu)
{
    // buf stride = 2048 bf16 elems = 4096 bytes for both tiles
    __shared__ __bf16 As[3][64 * 32];        // [buf][m][k], k contiguous
    __shared__ __bf16 Bs[3][4][32 * 16];     // [buf][ntile][n][k], k contiguous

    const int tid  = threadIdx.x;
    const int lane = tid & 31;
    const int wv   = tid >> 5;        // 0..7
    const int mt   = wv >> 1;         // 0..3  (16-row M subtile)
    const int nt   = (wv & 1) * 2;    // 0 or 2 (first of two 16-col N subtiles)
    const int m0b  = blockIdx.y * 64;
    const int n0b  = blockIdx.x * 64;

    // staging assignment: each thread owns 8 contiguous k of one row
    const int sr = tid >> 2;          // 0..63 row within tile
    const int sk = (tid & 3) * 8;     // k offset: 0,8,16,24
    const __bf16* Aptr = A + (long)(m0b + sr) * K + sk;
    const __bf16* Wptr = W + (long)(n0b + sr) * K + sk;

    // LDS raw byte offset of this thread's async destination in buffer 0
    const unsigned asBase = (unsigned)(size_t)&As[0][sr * 32 + sk];
    // B store base (buffer 0): Bs[0][sr>>4][(sr&15)*32 + sk]
    __bf16* const bsStore = &Bs[0][sr >> 4][0] + (sr & 15) * 32 + sk;
    // fragment read bases (buffer 0)
    const int kb = (lane >> 4) * 8;
    const int am = lane & 15;
    const __bf16* const aRead  = &As[0][(mt * 16 + am) * 32 + kb];
    const __bf16* const bRead0 = &Bs[0][nt][am * 32 + kb];
    const __bf16* const bRead1 = &Bs[0][nt + 1][am * 32 + kb];

    v8f acc0 = {}, acc1 = {};

    auto compute = [&](int buf) {
        const int e = buf * 2048;                  // elem offset per buffer
        v16bf_u a, b0, b1;
        a.h[0]  = *(const v8bf*)(aRead  + e);
        a.h[1]  = *(const v8bf*)(aRead  + e + 16);
        b0.h[0] = *(const v8bf*)(bRead0 + e);
        b0.h[1] = *(const v8bf*)(bRead0 + e + 16);
        b1.h[0] = *(const v8bf*)(bRead1 + e);
        b1.h[1] = *(const v8bf*)(bRead1 + e + 16);
        acc0 = __builtin_amdgcn_wmma_f32_16x16x32_bf16(
            false, a.v, false, b0.v, (short)0, acc0, false, false);
        acc1 = __builtin_amdgcn_wmma_f32_16x16x32_bf16(
            false, a.v, false, b1.v, (short)0, acc1, false, false);
    };

    // ---- prologue: stage 0 A-copy in flight, stage 0 B in registers ----
    asm volatile("global_load_async_to_lds_b128 %0, %1, off"
                 :: "v"(asBase), "v"((unsigned long long)(size_t)Aptr)
                 : "memory");
    v8bf breg = *(const v8bf*)Wptr;

    const int nsteps = K >> 5;        // >= 2 for all our shapes
    int buf = 0;
    for (int s = 0; s < nsteps - 1; ++s) {
        const int nb = (buf == 2) ? 0 : buf + 1;
        // issue next-stage A async copy (always valid here)
        asm volatile("global_load_async_to_lds_b128 %0, %1, off"
                     :: "v"(asBase + (unsigned)nb * 4096u),
                        "v"((unsigned long long)(size_t)(Aptr + (s + 1) * 32))
                     : "memory");
        // scatter current-stage B (no transpose, [n][k] with k contiguous) into LDS
        {
            __bf16* bp = bsStore + buf * 2048;
#pragma unroll
            for (int j = 0; j < 8; ++j) bp[j] = breg[j];
        }
        // prefetch next-stage B into registers
        breg = *(const v8bf*)(Wptr + (s + 1) * 32);
        // drain current stage's async copy (next stage stays in flight)
        asm volatile("s_wait_asynccnt 0x1" ::: "memory");
        __syncthreads();
        compute(buf);
        buf = nb;
    }
    // ---- peeled tail: stage nsteps-1 ----
    {
        __bf16* bp = bsStore + buf * 2048;
#pragma unroll
        for (int j = 0; j < 8; ++j) bp[j] = breg[j];
        asm volatile("s_wait_asynccnt 0x0" ::: "memory");
        __syncthreads();
        compute(buf);
    }

    // ---- epilogue (C/D layout: VGPR r -> M=r (lanes 0-15) / M=r+8 (lanes 16-31))
    const int m0    = m0b + mt * 16;
    const int n0    = n0b + nt * 16;
    const int nA    = n0 + (lane & 15);
    const int nB    = nA + 16;
    const int mbase = m0 + ((lane >> 4) ? 8 : 0);
    const float bvA = bias ? bias[nA] : 0.f;
    const float bvB = bias ? bias[nB] : 0.f;
#pragma unroll
    for (int r = 0; r < 8; ++r) {
        long row = (long)(mbase + r) * N;
        float v0 = acc0[r] + bvA;
        float v1 = acc1[r] + bvB;
        if (doGelu) {
            v0 = 0.5f * v0 * (1.f + erff(v0 * 0.7071067811865475f));
            v1 = 0.5f * v1 * (1.f + erff(v1 * 0.7071067811865475f));
        }
        if (resIn) { v0 += resIn[row + nA]; v1 += resIn[row + nB]; }
        if (outF)  { outF[row + nA] = v0;   outF[row + nB] = v1; }
        if (outB)  { outB[row + nA] = (__bf16)v0; outB[row + nB] = (__bf16)v1; }
    }
}

// ---------------------------------------------------------------------------
// Sliding-window attention: one wave32 per (b, t, h) query; 11-key band.
// qkv layout [B*T, 3D]: q | k | v, head h at offset h*64. Lane owns d, d+32.
// ---------------------------------------------------------------------------
__global__ __launch_bounds__(256) void attn_kernel(
    const __bf16* __restrict__ qkv,
    const unsigned char* __restrict__ pad,   // (B,T) bool
    __bf16* __restrict__ out)                // [B*T, D] bf16
{
    int qi   = blockIdx.x * 8 + (threadIdx.x >> 5);
    int lane = threadIdx.x & 31;
    int b    = qi / (Tc * Hc);
    int rem  = qi - b * Tc * Hc;
    int t    = rem / Hc;
    int h    = rem - t * Hc;

    const __bf16* qb = qkv + (long)(b * Tc + t) * (3 * Dc) + h * HDc;
    float q0 = (float)qb[lane]      * 0.125f;   // 1/sqrt(64)
    float q1 = (float)qb[lane + 32] * 0.125f;

    float s[11];
    float smax = -1e30f;
#pragma unroll
    for (int j = 0; j < 11; ++j) {
        int kt = t + j - 5;
        float sc = -1e30f;
        if (kt >= 0 && kt < Tc) {
            const __bf16* kb = qkv + (long)(b * Tc + kt) * (3 * Dc) + Dc + h * HDc;
            float d = q0 * (float)kb[lane] + q1 * (float)kb[lane + 32];
#pragma unroll
            for (int off = 16; off; off >>= 1) d += __shfl_xor(d, off, 32);
            sc = d;
            if (pad[b * Tc + kt]) sc -= 1e9f;
        }
        s[j] = sc;
        smax = fmaxf(smax, sc);
    }
    float sum = 0.f, o0 = 0.f, o1 = 0.f;
#pragma unroll
    for (int j = 0; j < 11; ++j) {
        int kt = t + j - 5;
        if (kt < 0 || kt >= Tc) continue;
        float e = __expf(s[j] - smax);
        sum += e;
        const __bf16* vb = qkv + (long)(b * Tc + kt) * (3 * Dc) + 2 * Dc + h * HDc;
        o0 += e * (float)vb[lane];
        o1 += e * (float)vb[lane + 32];
    }
    float inv = 1.f / sum;
    __bf16* op = out + (long)(b * Tc + t) * Dc + h * HDc;
    op[lane]      = (__bf16)(o0 * inv);
    op[lane + 32] = (__bf16)(o1 * inv);
}

// ---------------------------------------------------------------------------
// Host-side orchestration
// ---------------------------------------------------------------------------
extern "C" void kernel_launch(void* const* d_in, const int* in_sizes, int n_in,
                              void* d_out, int out_size, void* d_ws, size_t ws_size,
                              hipStream_t stream) {
    const float* x      = (const float*)d_in[0];
    const float* dw_w   = (const float*)d_in[1];
    const float* dw_b   = (const float*)d_in[2];
    const float* pw_w   = (const float*)d_in[3];
    const float* pw_b   = (const float*)d_in[4];
    const float* Wqkv   = (const float*)d_in[5];
    const float* bqkv   = (const float*)d_in[6];
    const float* Wo     = (const float*)d_in[7];
    const float* bo     = (const float*)d_in[8];
    const float* ln1_w  = (const float*)d_in[9];
    const float* ln1_b  = (const float*)d_in[10];
    const float* ln2_w  = (const float*)d_in[11];
    const float* ln2_b  = (const float*)d_in[12];
    const float* W1     = (const float*)d_in[13];
    const float* b1     = (const float*)d_in[14];
    const float* W2     = (const float*)d_in[15];
    const float* b2     = (const float*)d_in[16];
    const unsigned char* pad = (const unsigned char*)d_in[17];

    char* ws = (char*)d_ws;
    size_t off = 0;
    auto carve = [&](size_t bytes) {
        void* p = ws + off;
        off = (off + bytes + 255) & ~(size_t)255;
        return p;
    };
    const size_t nPW   = (size_t)Dc * Dc;            // 262144
    const size_t nQKV  = (size_t)Lc * 3 * Dc * Dc;   // 3145728
    const size_t nWo   = (size_t)Lc * Dc * Dc;       // 1048576
    const size_t nW1   = (size_t)Lc * 4 * Dc * Dc;   // 4194304
    const size_t nW2   = nW1;

    __bf16* wPW  = (__bf16*)carve(nPW  * 2);
    __bf16* wQKV = (__bf16*)carve(nQKV * 2);
    __bf16* wWo  = (__bf16*)carve(nWo  * 2);
    __bf16* wW1  = (__bf16*)carve(nW1  * 2);
    __bf16* wW2  = (__bf16*)carve(nW2  * 2);
    float*  xw   = (float*) carve((size_t)Mrows * Dc * 4);       // fp32 residual
    __bf16* ybf  = (__bf16*)carve((size_t)Mrows * Dc * 2);       // LN out / attn out
    __bf16* big  = (__bf16*)carve((size_t)Mrows * 4 * Dc * 2);   // dwconv/qkv/ffn-hid

    cvt_bf16_kernel<<<(int)(nPW  / 256), 256, 0, stream>>>(pw_w, wPW,  (long)nPW);
    cvt_bf16_kernel<<<(int)(nQKV / 256), 256, 0, stream>>>(Wqkv, wQKV, (long)nQKV);
    cvt_bf16_kernel<<<(int)(nWo  / 256), 256, 0, stream>>>(Wo,   wWo,  (long)nWo);
    cvt_bf16_kernel<<<(int)(nW1  / 256), 256, 0, stream>>>(W1,   wW1,  (long)nW1);
    cvt_bf16_kernel<<<(int)(nW2  / 256), 256, 0, stream>>>(W2,   wW2,  (long)nW2);

    const int elemBlocks = Mrows * Dc / 256;   // 24576
    dwconv_kernel<<<elemBlocks, 256, 0, stream>>>(x, dw_w, dw_b, big);
    gemm_bf16_kernel<<<dim3(Dc / 64, Mrows / 64), 256, 0, stream>>>(
        big, wPW, pw_b, x, xw, nullptr, Mrows, Dc, Dc, 1);
    pe_add_kernel<<<elemBlocks, 256, 0, stream>>>(xw);

    for (int l = 0; l < Lc; ++l) {
        ln_kernel<<<Mrows / 8, 256, 0, stream>>>(xw, ln1_w + l * Dc, ln1_b + l * Dc, ybf);
        gemm_bf16_kernel<<<dim3(3 * Dc / 64, Mrows / 64), 256, 0, stream>>>(
            ybf, wQKV + (size_t)l * 3 * Dc * Dc, bqkv + (size_t)l * 3 * Dc,
            nullptr, nullptr, big, Mrows, 3 * Dc, Dc, 0);
        attn_kernel<<<Mrows * Hc / 8, 256, 0, stream>>>(big, pad, ybf);
        gemm_bf16_kernel<<<dim3(Dc / 64, Mrows / 64), 256, 0, stream>>>(
            ybf, wWo + (size_t)l * Dc * Dc, bo + (size_t)l * Dc,
            xw, xw, nullptr, Mrows, Dc, Dc, 0);
        ln_kernel<<<Mrows / 8, 256, 0, stream>>>(xw, ln2_w + l * Dc, ln2_b + l * Dc, ybf);
        gemm_bf16_kernel<<<dim3(4 * Dc / 64, Mrows / 64), 256, 0, stream>>>(
            ybf, wW1 + (size_t)l * 4 * Dc * Dc, b1 + (size_t)l * 4 * Dc,
            nullptr, nullptr, big, Mrows, 4 * Dc, Dc, 1);
        gemm_bf16_kernel<<<dim3(Dc / 64, Mrows / 64), 256, 0, stream>>>(
            big, wW2 + (size_t)l * 4 * Dc * Dc, b2 + (size_t)l * Dc,
            xw, xw, nullptr, Mrows, Dc, 4 * Dc, 0);
    }

    hipMemcpyAsync(d_out, xw, (size_t)Mrows * Dc * sizeof(float),
                   hipMemcpyDeviceToDevice, stream);
}
